// ISTFTHead_61220463837702
// MI455X (gfx1250) — compile-verified, matrix operand order checked by reading, not run on hardware
//
#include <hip/hip_runtime.h>
#include <cstdint>
#include <cstddef>

// ---------------- problem constants ----------------
#define N_FFT    1024
#define HOP      256
#define DIM      512
#define BATCH    16
#define SEQ      4096
#define NBINS    513                    // n_fft/2 + 1
#define NOUT     1026                   // n_fft + 2
#define MPAD1    1088                   // padded output rows of GEMM1 (17 * 64)
#define KPAD2    1088                   // padded K of GEMM2 (17 * 64)
#define NCOLS    (BATCH * SEQ)          // 65536 flattened (b,l) columns
#define OUTLEN   ((SEQ - 1) * HOP + N_FFT)  // 1049344
#define PADC     ((N_FFT - HOP) / 2)    // 384 center-trim
#define OUT_PER_B (SEQ * HOP)           // 1048576
#define MAG_CLIPF 100.0f
#define TWO_PI   6.283185307179586476f

typedef __attribute__((ext_vector_type(16))) _Float16 v16h;
typedef __attribute__((ext_vector_type(8)))  _Float16 v8h;
typedef __attribute__((ext_vector_type(8)))  float    v8f;
typedef __attribute__((ext_vector_type(4)))  unsigned int u32x4;
typedef __attribute__((ext_vector_type(4)))  float    f32x4;

#define AS1 __attribute__((address_space(1)))
#define TO_G(T, p) ((AS1 T*)(uintptr_t)(p))
#define TO_CG(T, p) ((const AS1 T*)(uintptr_t)(p))

// ---------------- workspace layout (bytes) ----------------
constexpr size_t OFF_XT   = 0;                                           // f16 x^T [b*l][512]
constexpr size_t OFF_W16  = OFF_XT   + (size_t)BATCH * SEQ * DIM * 2;    // f16 W   [1088][512]
constexpr size_t OFF_BIAS = OFF_W16  + (size_t)MPAD1 * DIM * 2;          // f32 bias[1088]
constexpr size_t OFF_T16  = OFF_BIAS + 4608;                             // f16 T   [1024][1088]
constexpr size_t OFF_ENV  = OFF_T16  + (size_t)N_FFT * KPAD2 * 2;        // f32 1/env[OUTLEN]
constexpr size_t OFF_HT   = OFF_ENV  + (size_t)OUTLEN * 4;               // f32 h^T [b*l][1088]
constexpr size_t OFF_FRM  = OFF_HT;                                      // f16 frames[b*l][1024] (aliases hT)
constexpr size_t OFF_S16  = OFF_HT   + (size_t)NCOLS * MPAD1 * 4;        // f16 S^T [b*l][1088]

// ============================================================
// x (B, DIM, L) f32  ->  x^T f16 [b*L + l][DIM]   (LDS-tiled transpose)
// ============================================================
__global__ void k_xpose(const float* __restrict__ x, _Float16* __restrict__ xT) {
  __shared__ float tile[32][33];
  const int b  = blockIdx.z;
  const int l0 = blockIdx.x * 32;
  const int d0 = blockIdx.y * 32;
  const int tx = threadIdx.x, ty = threadIdx.y;
  const float* xb = x + (size_t)b * DIM * SEQ;
#pragma unroll
  for (int r = 0; r < 32; r += 8)
    tile[ty + r][tx] = xb[(size_t)(d0 + ty + r) * SEQ + (l0 + tx)];
  __syncthreads();
  _Float16* dst = xT + (size_t)b * SEQ * DIM;
#pragma unroll
  for (int r = 0; r < 32; r += 8)
    dst[(size_t)(l0 + ty + r) * DIM + (d0 + tx)] = (_Float16)tile[tx][ty + r];
}

// W (1026, 512) f32 -> f16 padded to 1088 rows (zeros)
__global__ void k_wcvt(const float* __restrict__ W, _Float16* __restrict__ W16) {
  const int idx = blockIdx.x * blockDim.x + threadIdx.x;
  if (idx >= MPAD1 * DIM) return;
  const int m = idx / DIM;
  W16[idx] = (m < NOUT) ? (_Float16)W[idx] : (_Float16)0.0f;
}

// bias (1026,) f32 -> padded 1088 f32
__global__ void k_bias(const float* __restrict__ b, float* __restrict__ biasPad) {
  const int m = blockIdx.x * blockDim.x + threadIdx.x;
  if (m >= MPAD1) return;
  biasPad[m] = (m < NOUT) ? b[m] : 0.0f;
}

// Windowed real-iDFT basis T[n][k2], n<1024, k2<1088 (f16).
__global__ void k_basis(_Float16* __restrict__ T16) {
  const int idx = blockIdx.x * blockDim.x + threadIdx.x;
  if (idx >= N_FFT * KPAD2) return;
  const int n  = idx / KPAD2;
  const int k2 = idx % KPAD2;
  const float win = 0.5f * (1.0f - cosf(TWO_PI * (float)n / (float)N_FFT));
  float val = 0.0f;
  if (k2 < NOUT) {
    const int  k    = (k2 < NBINS) ? k2 : (k2 - NBINS);
    const float coef = (k == 0 || k == N_FFT / 2) ? 1.0f : 2.0f;
    const float th   = TWO_PI * (float)((n * k) & (N_FFT - 1)) / (float)N_FFT;
    const float bfn  = (k2 < NBINS) ? cosf(th) : -sinf(th);
    val = coef * bfn * win * (1.0f / (float)N_FFT);
  }
  T16[idx] = (_Float16)val;
}

// exact window-square envelope reciprocal (<=4 overlapping frames)
__global__ void k_env(float* __restrict__ envInv) {
  const int u = blockIdx.x * blockDim.x + threadIdx.x;
  if (u >= OUTLEN) return;
  float e = 0.0f;
  const int lmax = u >> 8;
#pragma unroll
  for (int j = 0; j < 4; ++j) {
    const int l = lmax - j;
    if (l < 0 || l >= SEQ) continue;
    const int n = u - (l << 8);
    if (n < N_FFT) {
      const float w = 0.5f * (1.0f - cosf(TWO_PI * (float)n / (float)N_FFT));
      e += w * w;
    }
  }
  envInv[u] = (e > 1e-11f) ? (1.0f / e) : 1.0f;
}

// ============================================================
// f16 fragment via two global b128 loads (address space 1 forced so
// the backend emits global_load_b128, tracked only by LOADcnt).
// Per-lane base already includes row*ld + (lane>>4)*8.
// ============================================================
__device__ __forceinline__ v16h ld_frag(const AS1 _Float16* p) {
  struct Frag { u32x4 a, b; } f;
  f.a = *(const AS1 u32x4*)(p);
  f.b = *(const AS1 u32x4*)(p + 16);
  return __builtin_bit_cast(v16h, f);
}

#define MT 4   // 64 output rows per wave
#define NT 2   // 32 output cols per wave

// GEMM1: h^T[j][1088] = W16(1088x512) @ x^T(j,:)^T + bias   (f32 out)
__global__ void __launch_bounds__(256, 1)
k_gemm1(const _Float16* __restrict__ W16p, const _Float16* __restrict__ xTp,
        const float* __restrict__ biasPadp, float* __restrict__ hTp) {
  const AS1 _Float16* W16   = TO_CG(_Float16, W16p);
  const AS1 _Float16* xT    = TO_CG(_Float16, xTp);
  const AS1 float*   biasPad = TO_CG(float, biasPadp);
  AS1 float*         hT     = TO_G(float, hTp);

  const int wave = threadIdx.x >> 5;
  const int lane = threadIdx.x & 31;
  const int flat = blockIdx.x * 8 + wave;
  const int mBlocks = MPAD1 / (16 * MT);          // 17
  const int mb = flat % mBlocks;
  const int nb = flat / mBlocks;
  const int m0 = mb * 16 * MT;
  const int n0 = nb * 16 * NT;
  const int ln15 = lane & 15;
  const int hi8  = ((lane >> 4) & 1) << 3;

  const AS1 _Float16* pa[MT];
  const AS1 _Float16* pb[NT];
#pragma unroll
  for (int i = 0; i < MT; ++i) pa[i] = W16 + (size_t)(m0 + 16 * i + ln15) * DIM + hi8;
#pragma unroll
  for (int j = 0; j < NT; ++j) pb[j] = xT + (size_t)(n0 + 16 * j + ln15) * DIM + hi8;

  v8f acc[MT][NT] = {};
  v16h a0[MT], b0[NT], a1[MT], b1[NT];
#pragma unroll
  for (int i = 0; i < MT; ++i) a0[i] = ld_frag(pa[i]);
#pragma unroll
  for (int j = 0; j < NT; ++j) b0[j] = ld_frag(pb[j]);

  for (int k0 = 0; k0 < DIM - 64; k0 += 64) {
#pragma unroll
    for (int i = 0; i < MT; ++i) a1[i] = ld_frag(pa[i] + 32);
#pragma unroll
    for (int j = 0; j < NT; ++j) b1[j] = ld_frag(pb[j] + 32);
#pragma unroll
    for (int i = 0; i < MT; ++i)
#pragma unroll
      for (int j = 0; j < NT; ++j)
        acc[i][j] = __builtin_amdgcn_wmma_f32_16x16x32_f16(
            false, a0[i], false, b0[j], (short)0, acc[i][j], false, false);
#pragma unroll
    for (int i = 0; i < MT; ++i) { a0[i] = ld_frag(pa[i] + 64); pa[i] += 64; }
#pragma unroll
    for (int j = 0; j < NT; ++j) { b0[j] = ld_frag(pb[j] + 64); pb[j] += 64; }
#pragma unroll
    for (int i = 0; i < MT; ++i)
#pragma unroll
      for (int j = 0; j < NT; ++j)
        acc[i][j] = __builtin_amdgcn_wmma_f32_16x16x32_f16(
            false, a1[i], false, b1[j], (short)0, acc[i][j], false, false);
  }
  // tail 64
#pragma unroll
  for (int i = 0; i < MT; ++i) a1[i] = ld_frag(pa[i] + 32);
#pragma unroll
  for (int j = 0; j < NT; ++j) b1[j] = ld_frag(pb[j] + 32);
#pragma unroll
  for (int i = 0; i < MT; ++i)
#pragma unroll
    for (int j = 0; j < NT; ++j)
      acc[i][j] = __builtin_amdgcn_wmma_f32_16x16x32_f16(
          false, a0[i], false, b0[j], (short)0, acc[i][j], false, false);
#pragma unroll
  for (int i = 0; i < MT; ++i)
#pragma unroll
    for (int j = 0; j < NT; ++j)
      acc[i][j] = __builtin_amdgcn_wmma_f32_16x16x32_f16(
          false, a1[i], false, b1[j], (short)0, acc[i][j], false, false);

#pragma unroll
  for (int i = 0; i < MT; ++i) {
    const int mbase = m0 + 16 * i + hi8;
    const f32x4 bl = *(const AS1 f32x4*)(biasPad + mbase);
    const f32x4 bh = *(const AS1 f32x4*)(biasPad + mbase + 4);
#pragma unroll
    for (int j = 0; j < NT; ++j) {
      AS1 float* dst = hT + (size_t)(n0 + 16 * j + ln15) * MPAD1 + mbase;
      f32x4 o0 = {acc[i][j][0] + bl[0], acc[i][j][1] + bl[1],
                  acc[i][j][2] + bl[2], acc[i][j][3] + bl[3]};
      f32x4 o1 = {acc[i][j][4] + bh[0], acc[i][j][5] + bh[1],
                  acc[i][j][6] + bh[2], acc[i][j][7] + bh[3]};
      *(AS1 f32x4*)(dst)     = o0;
      *(AS1 f32x4*)(dst + 4) = o1;
    }
  }
}

// Elementwise: mag = min(exp(h),100); S^T[j][k]=mag*cos p, [513+k]=mag*sin p; zero pad
__global__ void k_epi(const float* __restrict__ hT, _Float16* __restrict__ S16) {
  const long long idx = (long long)blockIdx.x * blockDim.x + threadIdx.x;
  const int KE = NBINS + (KPAD2 - NOUT);          // 513 + 62 = 575
  if (idx >= (long long)NCOLS * KE) return;
  const int       k = (int)(idx % KE);
  const long long j = idx / KE;
  _Float16* srow = S16 + (size_t)j * KPAD2;
  if (k < NBINS) {
    const float* hrow = hT + (size_t)j * MPAD1;
    const float m = hrow[k];
    const float p = hrow[NBINS + k];
    const float mag = fminf(expf(m), MAG_CLIPF);
    srow[k]         = (_Float16)(mag * cosf(p));
    srow[NBINS + k] = (_Float16)(mag * sinf(p));
  } else {
    srow[NOUT + (k - NBINS)] = (_Float16)0.0f;    // K padding 1026..1087
  }
}

// GEMM2: frames[j][1024] = T16(1024x1088) @ S^T(j,:)^T   (f16 out)
__global__ void __launch_bounds__(256, 1)
k_gemm2(const _Float16* __restrict__ T16p, const _Float16* __restrict__ S16p,
        _Float16* __restrict__ framesp) {
  const AS1 _Float16* T16    = TO_CG(_Float16, T16p);
  const AS1 _Float16* S16    = TO_CG(_Float16, S16p);
  AS1 _Float16*       frames = TO_G(_Float16, framesp);

  const int wave = threadIdx.x >> 5;
  const int lane = threadIdx.x & 31;
  const int flat = blockIdx.x * 8 + wave;
  const int mBlocks = N_FFT / (16 * MT);          // 16
  const int mb = flat % mBlocks;
  const int nb = flat / mBlocks;
  const int m0 = mb * 16 * MT;
  const int n0 = nb * 16 * NT;
  const int ln15 = lane & 15;
  const int hi8  = ((lane >> 4) & 1) << 3;

  const AS1 _Float16* pa[MT];
  const AS1 _Float16* pb[NT];
#pragma unroll
  for (int i = 0; i < MT; ++i) pa[i] = T16 + (size_t)(m0 + 16 * i + ln15) * KPAD2 + hi8;
#pragma unroll
  for (int j = 0; j < NT; ++j) pb[j] = S16 + (size_t)(n0 + 16 * j + ln15) * KPAD2 + hi8;

  v8f acc[MT][NT] = {};
  v16h a0[MT], b0[NT], a1[MT], b1[NT];
#pragma unroll
  for (int i = 0; i < MT; ++i) a0[i] = ld_frag(pa[i]);
#pragma unroll
  for (int j = 0; j < NT; ++j) b0[j] = ld_frag(pb[j]);

  for (int k0 = 0; k0 < KPAD2 - 64; k0 += 64) {
#pragma unroll
    for (int i = 0; i < MT; ++i) a1[i] = ld_frag(pa[i] + 32);
#pragma unroll
    for (int j = 0; j < NT; ++j) b1[j] = ld_frag(pb[j] + 32);
#pragma unroll
    for (int i = 0; i < MT; ++i)
#pragma unroll
      for (int j = 0; j < NT; ++j)
        acc[i][j] = __builtin_amdgcn_wmma_f32_16x16x32_f16(
            false, a0[i], false, b0[j], (short)0, acc[i][j], false, false);
#pragma unroll
    for (int i = 0; i < MT; ++i) { a0[i] = ld_frag(pa[i] + 64); pa[i] += 64; }
#pragma unroll
    for (int j = 0; j < NT; ++j) { b0[j] = ld_frag(pb[j] + 64); pb[j] += 64; }
#pragma unroll
    for (int i = 0; i < MT; ++i)
#pragma unroll
      for (int j = 0; j < NT; ++j)
        acc[i][j] = __builtin_amdgcn_wmma_f32_16x16x32_f16(
            false, a1[i], false, b1[j], (short)0, acc[i][j], false, false);
  }
  // tail 64
#pragma unroll
  for (int i = 0; i < MT; ++i) a1[i] = ld_frag(pa[i] + 32);
#pragma unroll
  for (int j = 0; j < NT; ++j) b1[j] = ld_frag(pb[j] + 32);
#pragma unroll
  for (int i = 0; i < MT; ++i)
#pragma unroll
    for (int j = 0; j < NT; ++j)
      acc[i][j] = __builtin_amdgcn_wmma_f32_16x16x32_f16(
          false, a0[i], false, b0[j], (short)0, acc[i][j], false, false);
#pragma unroll
  for (int i = 0; i < MT; ++i)
#pragma unroll
    for (int j = 0; j < NT; ++j)
      acc[i][j] = __builtin_amdgcn_wmma_f32_16x16x32_f16(
          false, a1[i], false, b1[j], (short)0, acc[i][j], false, false);

#pragma unroll
  for (int i = 0; i < MT; ++i)
#pragma unroll
    for (int j = 0; j < NT; ++j) {
      v8h o;
#pragma unroll
      for (int r = 0; r < 8; ++r) o[r] = (_Float16)acc[i][j][r];
      *(AS1 v8h*)(frames + (size_t)(n0 + 16 * j + ln15) * N_FFT + m0 + 16 * i + hi8) = o;
    }
}

// Overlap-add (<=4 frames per sample) + envelope normalize + center trim
__global__ void k_ola(const _Float16* __restrict__ frames,
                      const float* __restrict__ envInv,
                      float* __restrict__ out) {
  const long long t = (long long)blockIdx.x * blockDim.x + threadIdx.x;
  if (t >= (long long)BATCH * OUT_PER_B) return;
  const int b  = (int)(t / OUT_PER_B);
  const int tt = (int)(t % OUT_PER_B);
  const int u  = tt + PADC;
  float acc = 0.0f;
  const int lmax = u >> 8;
#pragma unroll
  for (int j = 0; j < 4; ++j) {
    const int l = lmax - j;
    if (l < 0 || l >= SEQ) continue;
    const int n = u - (l << 8);
    acc += (float)frames[(size_t)(b * SEQ + l) * N_FFT + n];
  }
  out[t] = acc * envInv[u];
}

// ============================================================
extern "C" void kernel_launch(void* const* d_in, const int* in_sizes, int n_in,
                              void* d_out, int out_size, void* d_ws, size_t ws_size,
                              hipStream_t stream) {
  (void)in_sizes; (void)n_in; (void)out_size; (void)ws_size;
  const float* x    = (const float*)d_in[0];
  const float* W    = (const float*)d_in[1];
  const float* bias = (const float*)d_in[2];
  float* out = (float*)d_out;

  char* ws = (char*)d_ws;
  _Float16* xT16    = (_Float16*)(ws + OFF_XT);
  _Float16* W16     = (_Float16*)(ws + OFF_W16);
  float*    biasPad = (float*)   (ws + OFF_BIAS);
  _Float16* T16     = (_Float16*)(ws + OFF_T16);
  float*    envInv  = (float*)   (ws + OFF_ENV);
  float*    hT      = (float*)   (ws + OFF_HT);
  _Float16* frames  = (_Float16*)(ws + OFF_FRM);   // aliases hT (hT dead by then)
  _Float16* S16     = (_Float16*)(ws + OFF_S16);

  // prep
  k_xpose<<<dim3(SEQ / 32, DIM / 32, BATCH), dim3(32, 8), 0, stream>>>(x, xT16);
  k_wcvt <<<(MPAD1 * DIM + 255) / 256, 256, 0, stream>>>(W, W16);
  k_bias <<<(MPAD1 + 255) / 256, 256, 0, stream>>>(bias, biasPad);
  k_basis<<<(N_FFT * KPAD2 + 255) / 256, 256, 0, stream>>>(T16);
  k_env  <<<(OUTLEN + 255) / 256, 256, 0, stream>>>(envInv);

  // GEMM1: 17 m-blocks x (65536/32) n-blocks, 8 waves/workgroup
  k_gemm1<<<(17 * (NCOLS / 32)) / 8, 256, 0, stream>>>(W16, xT16, biasPad, hT);

  // nonlinearity -> S^T f16
  {
    const long long total = (long long)NCOLS * (NBINS + (KPAD2 - NOUT));
    k_epi<<<(unsigned)((total + 255) / 256), 256, 0, stream>>>(hT, S16);
  }

  // GEMM2: 16 m-blocks x (65536/32) n-blocks
  k_gemm2<<<(16 * (NCOLS / 32)) / 8, 256, 0, stream>>>(T16, S16, frames);

  // overlap-add + normalize
  {
    const long long total = (long long)BATCH * OUT_PER_B;
    k_ola<<<(unsigned)((total + 255) / 256), 256, 0, stream>>>(frames, envInv, out);
  }
}
